// Attention_28999619183000
// MI455X (gfx1250) — compile-verified, matrix operand order checked by reading, not run on hardware
//
#include <hip/hip_runtime.h>

// ---------------------------------------------------------------------------
// Types for CDNA5 WMMA
// ---------------------------------------------------------------------------
typedef __attribute__((ext_vector_type(16))) __bf16 v16bf;
typedef __attribute__((ext_vector_type(8)))  float  v8f;

#define HEADS    16
#define HEAD_DIM 64
#define DIM      1024
#define NTOK     1024   // 32*32
#define BATCH    16
#define INNER    1024   // HEADS*HEAD_DIM

// ---------------------------------------------------------------------------
// Cross-lane butterfly via ds_swizzle (wave32-safe, avoids shuffle headers)
// ---------------------------------------------------------------------------
template <int MASK>
__device__ inline float lane_xor(float v) {
    int iv = __builtin_amdgcn_ds_swizzle(__float_as_int(v), (MASK << 10) | 0x1f);
    return __int_as_float(iv);
}

// ---------------------------------------------------------------------------
// CDNA5 async global->LDS copy (ASYNCcnt-tracked DMA, bypasses VGPRs).
// lds_off: byte offset into LDS (low 32 bits of generic shared address).
// ---------------------------------------------------------------------------
__device__ inline void async_ld_b128(unsigned lds_off, const void* gaddr) {
    asm volatile("global_load_async_to_lds_b128 %0, %1, off"
                 :: "v"(lds_off), "v"(gaddr)
                 : "memory");
}
__device__ inline unsigned lds_off_of(const void* p) {
    return (unsigned)(unsigned long long)p;   // LDS aperture: addr[31:0] = LDS offset
}

// ---------------------------------------------------------------------------
// K0: mod = t @ mod_w^T   (16 x 2048)
// ---------------------------------------------------------------------------
__global__ __launch_bounds__(256) void mod_kernel(const float* __restrict__ t,
                                                  const float* __restrict__ mod_w,
                                                  float* __restrict__ mod) {
    int o = blockIdx.x * 256 + threadIdx.x;   // 0 .. 16*2048-1
    int b = o >> 11;
    int j = o & 2047;
    const float4* tr = reinterpret_cast<const float4*>(t + (size_t)b * DIM);
    const float4* wr = reinterpret_cast<const float4*>(mod_w + (size_t)j * DIM);
    float acc = 0.f;
    for (int i = 0; i < DIM / 4; ++i) {
        float4 a = tr[i], w = wr[i];
        acc += a.x * w.x + a.y * w.y + a.z * w.z + a.w * w.w;
    }
    mod[o] = acc;
}

// ---------------------------------------------------------------------------
// K1: f32 -> bf16 weight conversion
// ---------------------------------------------------------------------------
__global__ __launch_bounds__(256) void cvt_kernel(const float* __restrict__ src,
                                                  __bf16* __restrict__ dst, int n) {
    int i = blockIdx.x * 256 + threadIdx.x;
    if (i < n) dst[i] = (__bf16)src[i];
}

// ---------------------------------------------------------------------------
// K2: h = rmsnorm(x)*norm_w*(1+sc) + sh   -> bf16 (one block per row)
// ---------------------------------------------------------------------------
__global__ __launch_bounds__(256) void h_kernel(const float* __restrict__ x,
                                                const float* __restrict__ norm_w,
                                                const float* __restrict__ mod,
                                                __bf16* __restrict__ h) {
    const int row  = blockIdx.x;        // 0..16383
    const int b    = row >> 10;
    const int lane = threadIdx.x & 31;
    const int wave = threadIdx.x >> 5;
    const float* xr = x + (size_t)row * DIM;

    float ss = 0.f;
    for (int i = threadIdx.x; i < DIM; i += 256) { float v = xr[i]; ss += v * v; }
    ss += lane_xor<1>(ss);  ss += lane_xor<2>(ss);
    ss += lane_xor<4>(ss);  ss += lane_xor<8>(ss);  ss += lane_xor<16>(ss);

    __shared__ float red[8];
    if (lane == 0) red[wave] = ss;
    __syncthreads();
    float tot = 0.f;
    #pragma unroll
    for (int i = 0; i < 8; ++i) tot += red[i];
    float inv = rsqrtf(tot / (float)DIM + 1e-6f);

    const float* sc = mod + (size_t)b * 2 * DIM;
    const float* sh = sc + DIM;
    __bf16* hr = h + (size_t)row * DIM;
    for (int i = threadIdx.x; i < DIM; i += 256) {
        float v = xr[i] * inv * norm_w[i];
        v = v * (1.f + sc[i]) + sh[i];
        hr[i] = (__bf16)v;
    }
}

// ---------------------------------------------------------------------------
// K3: tiled bf16 WMMA GEMM:  C[M,N] = A[M,K] @ B[N,K]^T
//   mode 0: store f32 to Cout
//   mode 1: scatter qkv columns -> Q[b,h,n,d], K[b,h,n,d], VT[b,h,d,n] (bf16)
// Block tile 128x64, K-step 32, 8 waves, each wave 2x2 16x16 WMMA tiles.
// Double-buffered LDS fed by global_load_async_to_lds_b128 (ASYNCcnt).
// ---------------------------------------------------------------------------
#define BM 128
#define BN 64
#define BK 32

__global__ __launch_bounds__(256) void gemm_bf16_kernel(
        const __bf16* __restrict__ A, const __bf16* __restrict__ B,
        float* __restrict__ Cout,
        __bf16* __restrict__ Qo, __bf16* __restrict__ Ko, __bf16* __restrict__ VoT,
        int M, int N, int K, int mode)
{
    __shared__ __align__(32) __bf16 As[2][BM * BK];
    __shared__ __align__(32) __bf16 Bs[2][BN * BK];

    const int tid  = threadIdx.x;
    const int lane = tid & 31;
    const int wave = tid >> 5;
    const int hl   = lane & 15;   // position within 16-lane half
    const int hs   = lane >> 4;   // which half
    const int bm   = blockIdx.x * BM;
    const int bn   = blockIdx.y * BN;
    const int wr   = wave & 3;    // wave row  -> m offset wr*32
    const int wc   = wave >> 2;   // wave col  -> n offset wc*32

    // per-thread async copy slots
    const int aidx = tid * 16;                 // A tile: 16 bf16 = 2 x b128
    const int ar = aidx >> 5, ac = aidx & 31;
    const int bidx = tid * 8;                  // B tile: 8 bf16 = 1 x b128
    const int br = bidx >> 5, bc = bidx & 31;

    const __bf16* gA = A + (size_t)(bm + ar) * K + ac;
    const __bf16* gB = B + (size_t)(bn + br) * K + bc;

    auto issue_tile = [&](int buf, int k0) {
        async_ld_b128(lds_off_of(&As[buf][aidx]),     gA + k0);
        async_ld_b128(lds_off_of(&As[buf][aidx + 8]), gA + k0 + 8);
        async_ld_b128(lds_off_of(&Bs[buf][bidx]),     gB + k0);
    };

    v8f c00 = {}, c01 = {}, c10 = {}, c11 = {};

    const int ksteps = K / BK;
    issue_tile(0, 0);                              // prologue

    for (int kt = 0; kt < ksteps; ++kt) {
        const int cur = kt & 1;
        const bool more = (kt + 1) < ksteps;
        if (more) issue_tile(cur ^ 1, (kt + 1) * BK);

        // wait for the *older* batch (3 async ops/wave per tile batch)
        if (more) asm volatile("s_wait_asynccnt 0x3" ::: "memory");
        else      asm volatile("s_wait_asynccnt 0x0" ::: "memory");
        __syncthreads();

        v16bf a0 = *reinterpret_cast<const v16bf*>(&As[cur][(wr * 32 +      hl) * BK + hs * 16]);
        v16bf a1 = *reinterpret_cast<const v16bf*>(&As[cur][(wr * 32 + 16 + hl) * BK + hs * 16]);
        v16bf b0 = *reinterpret_cast<const v16bf*>(&Bs[cur][(wc * 32 +      hl) * BK + hs * 16]);
        v16bf b1 = *reinterpret_cast<const v16bf*>(&Bs[cur][(wc * 32 + 16 + hl) * BK + hs * 16]);

        c00 = __builtin_amdgcn_wmma_f32_16x16x32_bf16(false, a0, false, b0, (short)0, c00, false, false);
        c01 = __builtin_amdgcn_wmma_f32_16x16x32_bf16(false, a0, false, b1, (short)0, c01, false, false);
        c10 = __builtin_amdgcn_wmma_f32_16x16x32_bf16(false, a1, false, b0, (short)0, c10, false, false);
        c11 = __builtin_amdgcn_wmma_f32_16x16x32_bf16(false, a1, false, b1, (short)0, c11, false, false);
        __syncthreads();   // readers done before next overwrite of this buffer
    }

    // Epilogue. C 16x16 f32 layout: vgpr i -> row i + 8*(lane>=16), col = lane&15.
    #pragma unroll
    for (int ti = 0; ti < 2; ++ti) {
        #pragma unroll
        for (int tj = 0; tj < 2; ++tj) {
            v8f acc = (ti == 0) ? (tj == 0 ? c00 : c01) : (tj == 0 ? c10 : c11);
            int rbase = bm + wr * 32 + ti * 16;
            int cbase = bn + wc * 32 + tj * 16;
            #pragma unroll
            for (int i = 0; i < 8; ++i) {
                int grow = rbase + i + hs * 8;
                int gcol = cbase + hl;
                float v = acc[i];
                if (mode == 0) {
                    Cout[(size_t)grow * N + gcol] = v;
                } else {
                    // qkv scatter: col -> (head, part, within)
                    int b    = grow >> 10;
                    int nrow = grow & 1023;
                    int head = gcol / 192;
                    int rr   = gcol - head * 192;
                    int part = rr >> 6;
                    int wd   = rr & 63;
                    int bh   = b * HEADS + head;
                    if (part == 0)
                        Qo[((size_t)bh * NTOK + nrow) * HEAD_DIM + wd] = (__bf16)v;
                    else if (part == 1)
                        Ko[((size_t)bh * NTOK + nrow) * HEAD_DIM + wd] = (__bf16)v;
                    else
                        VoT[((size_t)bh * HEAD_DIM + wd) * NTOK + nrow] = (__bf16)v;
                }
            }
        }
    }
}

// ---------------------------------------------------------------------------
// K4: in-place 2D RoPE over [bh, n, 64] (+ optional q scaling)
// pair p (0..31): p<16 -> x_pos * theta[p], else y_pos * theta[p-16]
// ---------------------------------------------------------------------------
__global__ __launch_bounds__(256) void rope_kernel(__bf16* __restrict__ buf, float scale) {
    int idx = blockIdx.x * 256 + threadIdx.x;   // bh*1024*32 pairs
    int p  = idx & 31;
    int n  = (idx >> 5) & 1023;
    int bh = idx >> 15;
    float pos   = (p < 16) ? (float)(n & 31) : (float)(n >> 5);
    int   pj    = p & 15;
    float theta = __powf(10000.f, -(float)pj / 16.f);
    float ang   = pos * theta;
    float cs = __cosf(ang), sn = __sinf(ang);
    size_t base = (((size_t)bh * NTOK) + n) * HEAD_DIM + 2 * p;
    float x0 = (float)buf[base], x1 = (float)buf[base + 1];
    buf[base]     = (__bf16)((x0 * cs - x1 * sn) * scale);
    buf[base + 1] = (__bf16)((x1 * cs + x0 * sn) * scale);
}

// ---------------------------------------------------------------------------
// K5: flash attention. block = 4 waves; wave handles 16 q rows of one (b,h).
// Q,K in [bh,n,64] bf16 (q pre-scaled), VT in [bh,64,n] bf16.
// Output O in [b, n, h*64+d] bf16 (row-major M=16384, K=1024 for final GEMM).
// ---------------------------------------------------------------------------
__global__ __launch_bounds__(128) void attn_kernel(
        const __bf16* __restrict__ Q, const __bf16* __restrict__ Kb,
        const __bf16* __restrict__ VT, __bf16* __restrict__ O)
{
    const int lane = threadIdx.x & 31;
    const int wave = threadIdx.x >> 5;
    const int hl   = lane & 15;
    const int hs   = lane >> 4;
    const int bh   = blockIdx.x;            // 0..255
    const int b    = bh >> 4;
    const int hh   = bh & 15;
    const int q0   = blockIdx.y * 64 + wave * 16;

    __shared__ __align__(32) __bf16 Pl[4][16][32];

    const __bf16* qb = Q  + (((size_t)bh * NTOK) + q0) * HEAD_DIM;
    const __bf16* kb = Kb + (size_t)bh * NTOK * HEAD_DIM;
    const __bf16* vb = VT + (size_t)bh * HEAD_DIM * NTOK;

    // Q fragments (16x32 A-tiles over d)
    v16bf aq0 = *reinterpret_cast<const v16bf*>(qb + (size_t)hl * HEAD_DIM + hs * 16);
    v16bf aq1 = *reinterpret_cast<const v16bf*>(qb + (size_t)hl * HEAD_DIM + hs * 16 + 32);

    v8f o0 = {}, o1 = {}, o2 = {}, o3 = {};
    float m_r[8], l_r[8];
    #pragma unroll
    for (int i = 0; i < 8; ++i) { m_r[i] = -1e30f; l_r[i] = 0.f; }

    for (int n0 = 0; n0 < NTOK; n0 += 32) {
        // ---- S = q @ k^T over this 32-key slab (two 16x16 tiles) ----
        v16bf bk00 = *reinterpret_cast<const v16bf*>(kb + (size_t)(n0 +      hl) * HEAD_DIM + hs * 16);
        v16bf bk01 = *reinterpret_cast<const v16bf*>(kb + (size_t)(n0 +      hl) * HEAD_DIM + hs * 16 + 32);
        v16bf bk10 = *reinterpret_cast<const v16bf*>(kb + (size_t)(n0 + 16 + hl) * HEAD_DIM + hs * 16);
        v16bf bk11 = *reinterpret_cast<const v16bf*>(kb + (size_t)(n0 + 16 + hl) * HEAD_DIM + hs * 16 + 32);
        v8f s0 = {}, s1 = {};
        s0 = __builtin_amdgcn_wmma_f32_16x16x32_bf16(false, aq0, false, bk00, (short)0, s0, false, false);
        s0 = __builtin_amdgcn_wmma_f32_16x16x32_bf16(false, aq1, false, bk01, (short)0, s0, false, false);
        s1 = __builtin_amdgcn_wmma_f32_16x16x32_bf16(false, aq0, false, bk10, (short)0, s1, false, false);
        s1 = __builtin_amdgcn_wmma_f32_16x16x32_bf16(false, aq1, false, bk11, (short)0, s1, false, false);

        // ---- online softmax (row-wise; reduce across 16 lanes via ds_swizzle) ----
        float alpha[8];
        #pragma unroll
        for (int i = 0; i < 8; ++i) {
            float mx = fmaxf(s0[i], s1[i]);
            mx = fmaxf(mx, lane_xor<1>(mx));
            mx = fmaxf(mx, lane_xor<2>(mx));
            mx = fmaxf(mx, lane_xor<4>(mx));
            mx = fmaxf(mx, lane_xor<8>(mx));
            float mnew = fmaxf(m_r[i], mx);
            alpha[i]   = __expf(m_r[i] - mnew);
            float p0   = __expf(s0[i] - mnew);
            float p1   = __expf(s1[i] - mnew);
            float ts   = p0 + p1;
            ts += lane_xor<1>(ts); ts += lane_xor<2>(ts);
            ts += lane_xor<4>(ts); ts += lane_xor<8>(ts);
            l_r[i] = l_r[i] * alpha[i] + ts;
            m_r[i] = mnew;
            int row = i + hs * 8;
            Pl[wave][row][hl]      = (__bf16)p0;
            Pl[wave][row][hl + 16] = (__bf16)p1;
        }
        __syncthreads();

        // ---- O = O*alpha + P @ V  (P as 16x32 A-tile, VT rows as B) ----
        v16bf pA  = *reinterpret_cast<const v16bf*>(&Pl[wave][hl][hs * 16]);
        v16bf bv0 = *reinterpret_cast<const v16bf*>(vb + (size_t)(      hl) * NTOK + n0 + hs * 16);
        v16bf bv1 = *reinterpret_cast<const v16bf*>(vb + (size_t)(16 + hl) * NTOK + n0 + hs * 16);
        v16bf bv2 = *reinterpret_cast<const v16bf*>(vb + (size_t)(32 + hl) * NTOK + n0 + hs * 16);
        v16bf bv3 = *reinterpret_cast<const v16bf*>(vb + (size_t)(48 + hl) * NTOK + n0 + hs * 16);
        #pragma unroll
        for (int i = 0; i < 8; ++i) {
            o0[i] *= alpha[i]; o1[i] *= alpha[i];
            o2[i] *= alpha[i]; o3[i] *= alpha[i];
        }
        o0 = __builtin_amdgcn_wmma_f32_16x16x32_bf16(false, pA, false, bv0, (short)0, o0, false, false);
        o1 = __builtin_amdgcn_wmma_f32_16x16x32_bf16(false, pA, false, bv1, (short)0, o1, false, false);
        o2 = __builtin_amdgcn_wmma_f32_16x16x32_bf16(false, pA, false, bv2, (short)0, o2, false, false);
        o3 = __builtin_amdgcn_wmma_f32_16x16x32_bf16(false, pA, false, bv3, (short)0, o3, false, false);
        __syncthreads();
    }

    // ---- normalize + store [b, n, h*64 + d] ----
    #pragma unroll
    for (int i = 0; i < 8; ++i) {
        int row = q0 + i + hs * 8;
        float inv = 1.f / l_r[i];
        size_t base = ((size_t)(b * NTOK + row)) * INNER + hh * HEAD_DIM;
        O[base +      hl] = (__bf16)(o0[i] * inv);
        O[base + 16 + hl] = (__bf16)(o1[i] * inv);
        O[base + 32 + hl] = (__bf16)(o2[i] * inv);
        O[base + 48 + hl] = (__bf16)(o3[i] * inv);
    }
}

// ---------------------------------------------------------------------------
// Host-side launch
// ---------------------------------------------------------------------------
extern "C" void kernel_launch(void* const* d_in, const int* in_sizes, int n_in,
                              void* d_out, int out_size, void* d_ws, size_t ws_size,
                              hipStream_t stream) {
    const float* x      = (const float*)d_in[0];
    const float* t      = (const float*)d_in[1];
    const float* norm_w = (const float*)d_in[2];
    const float* mod_w  = (const float*)d_in[3];
    const float* qkv_w  = (const float*)d_in[4];
    const float* wo_w   = (const float*)d_in[5];
    float* out = (float*)d_out;

    char* ws = (char*)d_ws;
    size_t off = 0;
    auto alloc = [&](size_t bytes) { char* p = ws + off; off = (off + bytes + 255) & ~(size_t)255; return p; };

    float*  mod      = (float*)  alloc((size_t)BATCH * 2 * DIM * sizeof(float));
    __bf16* qkvw_bf  = (__bf16*) alloc((size_t)3 * INNER * DIM * sizeof(__bf16));
    __bf16* wow_bf   = (__bf16*) alloc((size_t)DIM * INNER * sizeof(__bf16));
    __bf16* hbuf     = (__bf16*) alloc((size_t)BATCH * NTOK * DIM * sizeof(__bf16));
    __bf16* Qbuf     = (__bf16*) alloc((size_t)BATCH * HEADS * NTOK * HEAD_DIM * sizeof(__bf16));
    __bf16* Kbuf     = (__bf16*) alloc((size_t)BATCH * HEADS * NTOK * HEAD_DIM * sizeof(__bf16));
    __bf16* VTbuf    = (__bf16*) alloc((size_t)BATCH * HEADS * HEAD_DIM * NTOK * sizeof(__bf16));
    __bf16* AObuf    = (__bf16*) alloc((size_t)BATCH * NTOK * INNER * sizeof(__bf16));
    (void)ws_size; (void)in_sizes; (void)n_in; (void)out_size;

    // 1) modulation vectors
    mod_kernel<<<(BATCH * 2 * DIM) / 256, 256, 0, stream>>>(t, mod_w, mod);

    // 2) weight downconversion
    {
        int n1 = 3 * INNER * DIM;
        cvt_kernel<<<(n1 + 255) / 256, 256, 0, stream>>>(qkv_w, qkvw_bf, n1);
        int n2 = DIM * INNER;
        cvt_kernel<<<(n2 + 255) / 256, 256, 0, stream>>>(wo_w, wow_bf, n2);
    }

    // 3) rmsnorm + modulation -> h (bf16)
    h_kernel<<<BATCH * NTOK, 256, 0, stream>>>(x, norm_w, mod, hbuf);

    // 4) qkv = h @ qkv_w^T with scatter epilogue (Q/K [bh,n,d], V transposed)
    {
        dim3 grid((BATCH * NTOK) / BM, (3 * INNER) / BN);
        gemm_bf16_kernel<<<grid, 256, 0, stream>>>(hbuf, qkvw_bf, nullptr,
                                                   Qbuf, Kbuf, VTbuf,
                                                   BATCH * NTOK, 3 * INNER, DIM, 1);
    }

    // 5) RoPE (+ softmax scale folded into q)
    {
        int pairs = BATCH * HEADS * NTOK * 32;
        rope_kernel<<<pairs / 256, 256, 0, stream>>>(Qbuf, 0.125f);  // 1/sqrt(64)
        rope_kernel<<<pairs / 256, 256, 0, stream>>>(Kbuf, 1.0f);
    }

    // 6) flash attention
    {
        dim3 grid(BATCH * HEADS, NTOK / 64);
        attn_kernel<<<grid, 128, 0, stream>>>(Qbuf, Kbuf, VTbuf, AObuf);
    }

    // 7) out = attn_out @ wo_w^T  (fp32 result)
    {
        dim3 grid((BATCH * NTOK) / BM, INNER / BN);
        gemm_bf16_kernel<<<grid, 256, 0, stream>>>(AObuf, wow_bf, out,
                                                   nullptr, nullptr, nullptr,
                                                   BATCH * NTOK, INNER, DIM, 0);
    }
}
